// LSTM_35433480192802
// MI455X (gfx1250) — matrix operand order repaired and checked by reading, not compiled, not run on hardware
//
#include <hip/hip_runtime.h>
#include <hip/hip_bf16.h>

// Pipeline: MLP(8->64->128->4->1, LeakyReLU) -> scalar LSTM scan (SEQ=1024)
// -> [8192,1024] x [1024,1024]^T GEMM via bf16x3 split-precision WMMA.
// Workspace layout (requires ~36.1 MB):
//   [0,32KB)              : z  (MLP output, 8192 f32)
//   [32KB, 32KB+32MB)     : H  (LSTM hidden states, 8192x1024 f32, row-major)
//   [+0, +2MB)            : Wout_hi (1024x1024 bf16)
//   [+2MB, +4MB)          : Wout_lo (1024x1024 bf16)

#define B_DIM 8192
#define SEQ   1024

typedef __attribute__((ext_vector_type(16))) __bf16 bf16x16;
typedef __attribute__((ext_vector_type(8)))  float  f32x8;

// ---------------- fast transcendentals (short dependent chains) -------------
__device__ __forceinline__ float sigmoid_f(float x) {
#if __has_builtin(__builtin_amdgcn_exp2f) && __has_builtin(__builtin_amdgcn_rcpf)
  return __builtin_amdgcn_rcpf(1.0f + __builtin_amdgcn_exp2f(-1.44269504088896f * x));
#else
  return 1.0f / (1.0f + __expf(-x));
#endif
}
__device__ __forceinline__ float tanh_f(float x) {
#if __has_builtin(__builtin_amdgcn_tanhf)
  return __builtin_amdgcn_tanhf(x);   // gfx1250 native v_tanh_f32
#else
  return fmaf(2.0f, sigmoid_f(2.0f * x), -1.0f);
#endif
}
__device__ __forceinline__ float leaky(float x) { return x > 0.0f ? x : 0.2f * x; }

// ---------------- Kernel 1: per-row MLP, weights staged in LDS --------------
__global__ __launch_bounds__(256) void mlp_kernel(
    const float* __restrict__ x,
    const float* __restrict__ W1, const float* __restrict__ b1,
    const float* __restrict__ W2, const float* __restrict__ b2,
    const float* __restrict__ W3, const float* __restrict__ b3,
    const float* __restrict__ W4, const float* __restrict__ b4,
    float* __restrict__ z)
{
  __shared__ float sW1[64 * 8], sb1[64];
  __shared__ float sW2[128 * 64], sb2[128];
  __shared__ float sW3[4 * 128], sb3[4], sW4[4];
  const int tid = threadIdx.x;
  for (int i = tid; i < 64 * 8;   i += 256) sW1[i] = W1[i];
  for (int i = tid; i < 64;       i += 256) sb1[i] = b1[i];
  for (int i = tid; i < 128 * 64; i += 256) sW2[i] = W2[i];
  for (int i = tid; i < 128;      i += 256) sb2[i] = b2[i];
  for (int i = tid; i < 4 * 128;  i += 256) sW3[i] = W3[i];
  if (tid < 4) { sb3[tid] = b3[tid]; sW4[tid] = W4[tid]; }
  __syncthreads();

  const int b = blockIdx.x * 256 + tid;
  float xin[8];
#pragma unroll
  for (int k = 0; k < 8; ++k) xin[k] = x[(size_t)b * 8 + k];

  float z1[64];
#pragma unroll
  for (int j = 0; j < 64; ++j) {          // fully unrolled -> z1 stays in VGPRs
    float a = sb1[j];
#pragma unroll
    for (int k = 0; k < 8; ++k) a = fmaf(sW1[j * 8 + k], xin[k], a);
    z1[j] = leaky(a);
  }

  float z3[4] = {sb3[0], sb3[1], sb3[2], sb3[3]};
  for (int j = 0; j < 128; ++j) {         // rolled outer, unrolled inner
    float a = sb2[j];
#pragma unroll
    for (int k = 0; k < 64; ++k) a = fmaf(sW2[j * 64 + k], z1[k], a);
    a = leaky(a);
#pragma unroll
    for (int m = 0; m < 4; ++m) z3[m] = fmaf(sW3[m * 128 + j], a, z3[m]);
  }

  float a4 = b4[0];
#pragma unroll
  for (int m = 0; m < 4; ++m) a4 = fmaf(sW4[m], leaky(z3[m]), a4);
  z[b] = leaky(a4);
}

// ---------------- Kernel 2: latency-bound LSTM scan -------------------------
__global__ __launch_bounds__(256) void lstm_kernel(
    const float* __restrict__ z, const float* __restrict__ h0v,
    const float* __restrict__ c0v,
    const float* __restrict__ W_ih, const float* __restrict__ b_ih,
    const float* __restrict__ W_hh, const float* __restrict__ b_hh,
    float* __restrict__ H)
{
  const int b = blockIdx.x * blockDim.x + threadIdx.x;
  const float wi0 = W_ih[0], wi1 = W_ih[1], wi2 = W_ih[2], wi3 = W_ih[3];
  const float wh0 = W_hh[0], wh1 = W_hh[1], wh2 = W_hh[2], wh3 = W_hh[3];
  const float bb0 = b_ih[0] + b_hh[0], bb1 = b_ih[1] + b_hh[1];
  const float bb2 = b_ih[2] + b_hh[2], bb3 = b_ih[3] + b_hh[3];

  float xin = z[b], h = h0v[b], c = c0v[b];
  float* __restrict__ row = H + (size_t)b * SEQ;

  // step 0: input (MLP output) differs from hidden state
  {
    const float gi = fmaf(xin, wi0, fmaf(h, wh0, bb0));
    const float gf = fmaf(xin, wi1, fmaf(h, wh1, bb1));
    const float gg = fmaf(xin, wi2, fmaf(h, wh2, bb2));
    const float go = fmaf(xin, wi3, fmaf(h, wh3, bb3));
    c = fmaf(sigmoid_f(gf), c, sigmoid_f(gi) * tanh_f(gg));
    h = sigmoid_f(go) * tanh_f(c);
    row[0] = h;
  }
  // steps 1..SEQ-1: torch code feeds h back as input -> fold weights
  const float wc0 = wi0 + wh0, wc1 = wi1 + wh1, wc2 = wi2 + wh2, wc3 = wi3 + wh3;
  for (int t = 1; t < SEQ; ++t) {
    const float gi = fmaf(h, wc0, bb0);
    const float gf = fmaf(h, wc1, bb1);
    const float gg = fmaf(h, wc2, bb2);
    const float go = fmaf(h, wc3, bb3);
    c = fmaf(sigmoid_f(gf), c, sigmoid_f(gi) * tanh_f(gg));
    h = sigmoid_f(go) * tanh_f(c);
    row[t] = h;
  }
}

// ---------------- Kernel 3: split Wout into bf16 hi/lo ----------------------
__global__ __launch_bounds__(256) void split_kernel(
    const float* __restrict__ W, __bf16* __restrict__ hi, __bf16* __restrict__ lo)
{
  const int i = blockIdx.x * blockDim.x + threadIdx.x;
  const float w = W[i];
  const __bf16 h = (__bf16)w;
  hi[i] = h;
  lo[i] = (__bf16)(w - (float)h);
}

// ---------------- Kernel 4: bf16x3 split-precision WMMA GEMM ----------------
// out[b,s] = sum_k H[b,k] * Wout[s,k] + bout[s]
// Wave = 2 M-tiles x 8 N-tiles; block = 4 waves (128 rows) x 128 cols.
__global__ __launch_bounds__(128) void gemm_wmma_kernel(
    const float* __restrict__ H, const __bf16* __restrict__ Whi,
    const __bf16* __restrict__ Wlo, const float* __restrict__ bout,
    float* __restrict__ out)
{
  const int lane  = threadIdx.x & 31;
  const int wave  = threadIdx.x >> 5;
  const int mBase = blockIdx.x * 128 + wave * 32;   // 2 M-tiles per wave
  const int nBase = blockIdx.y * 128;               // 8 N-tiles per wave
  const int l15   = lane & 15;
  const int hi16  = lane >> 4;                      // 0: lanes 0-15, 1: 16-31

  // C/D layout: VGPR r holds M=r (lanes 0-15) / M=r+8 (lanes 16-31), N=l15.
  f32x8 acc[2][8];
#pragma unroll
  for (int t = 0; t < 8; ++t) {
    const float bv = bout[nBase + t * 16 + l15];    // same col for all 8 rows
#pragma unroll
    for (int u = 0; u < 2; ++u)
#pragma unroll
      for (int r = 0; r < 8; ++r) acc[u][t][r] = bv;
  }

  const float* arow0 = H + (size_t)(mBase + l15) * SEQ;
  const float* arow1 = H + (size_t)(mBase + 16 + l15) * SEQ;

  for (int k0 = 0; k0 < SEQ; k0 += 32) {
    // A k order used here (contiguous, same as B): lanes 0-15 hold K=0..15, lanes 16-31 K=16..31.
    // Convert f32 H -> bf16 hi/lo on the fly (co-executes with XDL WMMA).
    bf16x16 ahi[2], alo[2];
#pragma unroll
    for (int u = 0; u < 2; ++u) {
      const float* p0 = (u ? arow1 : arow0) + k0 + 16 * hi16;
      const float* p1 = (u ? arow1 : arow0) + k0 + 8 + 16 * hi16;
#pragma unroll
      for (int i = 0; i < 8; ++i) {
        const float v0 = p0[i], v1 = p1[i];
        const __bf16 c0 = (__bf16)v0, c1 = (__bf16)v1;
        ahi[u][i]     = c0;  alo[u][i]     = (__bf16)(v0 - (float)c0);
        ahi[u][i + 8] = c1;  alo[u][i + 8] = (__bf16)(v1 - (float)c1);
      }
    }
#pragma unroll
    for (int t = 0; t < 8; ++t) {
      // B k order used here (matches A): lane = N, element i -> K = i + 16*hi16;
      // per-lane 16 contiguous bf16 = one aligned 32B load.
      const size_t wOff = (size_t)(nBase + t * 16 + l15) * SEQ + k0 + 16 * hi16;
      const bf16x16 bhi = *(const bf16x16*)(Whi + wOff);
      const bf16x16 blo = *(const bf16x16*)(Wlo + wOff);
#pragma unroll
      for (int u = 0; u < 2; ++u) {
        acc[u][t] = __builtin_amdgcn_wmma_f32_16x16x32_bf16(
            false, ahi[u], false, bhi, (short)0, acc[u][t], false, false);
        acc[u][t] = __builtin_amdgcn_wmma_f32_16x16x32_bf16(
            false, ahi[u], false, blo, (short)0, acc[u][t], false, false);
        acc[u][t] = __builtin_amdgcn_wmma_f32_16x16x32_bf16(
            false, alo[u], false, bhi, (short)0, acc[u][t], false, false);
      }
    }
  }

#pragma unroll
  for (int u = 0; u < 2; ++u)
#pragma unroll
    for (int t = 0; t < 8; ++t) {
      const int col = nBase + t * 16 + l15;
#pragma unroll
      for (int r = 0; r < 8; ++r)
        out[(size_t)(mBase + u * 16 + r + 8 * hi16) * SEQ + col] = acc[u][t][r];
    }
}

// ---------------- launch -----------------------------------------------------
extern "C" void kernel_launch(void* const* d_in, const int* in_sizes, int n_in,
                              void* d_out, int out_size, void* d_ws, size_t ws_size,
                              hipStream_t stream) {
  (void)in_sizes; (void)n_in; (void)out_size; (void)ws_size;
  const float* x    = (const float*)d_in[0];
  const float* h0   = (const float*)d_in[1];
  const float* c0   = (const float*)d_in[2];
  const float* W1   = (const float*)d_in[3];
  const float* b1   = (const float*)d_in[4];
  const float* W2   = (const float*)d_in[5];
  const float* b2   = (const float*)d_in[6];
  const float* W3   = (const float*)d_in[7];
  const float* b3   = (const float*)d_in[8];
  const float* W4   = (const float*)d_in[9];
  const float* b4   = (const float*)d_in[10];
  const float* W_ih = (const float*)d_in[11];
  const float* b_ih = (const float*)d_in[12];
  const float* W_hh = (const float*)d_in[13];
  const float* b_hh = (const float*)d_in[14];
  const float* Wout = (const float*)d_in[15];
  const float* bout = (const float*)d_in[16];

  char* ws = (char*)d_ws;
  float*  z   = (float*)ws;                                        // 32 KB
  float*  H   = (float*)(ws + (32ull << 10));                      // 32 MB
  __bf16* Whi = (__bf16*)(ws + (32ull << 10) + (32ull << 20));     // 2 MB
  __bf16* Wlo = Whi + (size_t)SEQ * SEQ;                           // 2 MB

  mlp_kernel<<<B_DIM / 256, 256, 0, stream>>>(x, W1, b1, W2, b2, W3, b3, W4, b4, z);
  lstm_kernel<<<B_DIM / 256, 256, 0, stream>>>(z, h0, c0, W_ih, b_ih, W_hh, b_hh, H);
  split_kernel<<<(SEQ * SEQ) / 256, 256, 0, stream>>>(Wout, Whi, Wlo);
  dim3 grid(B_DIM / 128, SEQ / 128);
  gemm_wmma_kernel<<<grid, 128, 0, stream>>>(H, Whi, Wlo, bout, (float*)d_out);
}
